// Decoder_25013889532483
// MI455X (gfx1250) — matrix operand order repaired and checked by reading, not compiled, not run on hardware
//
#include <hip/hip_runtime.h>
#include <hip/hip_bf16.h>

typedef __attribute__((ext_vector_type(16))) __bf16 v16bf;
typedef __attribute__((ext_vector_type(8)))  __bf16 v8bf;
typedef __attribute__((ext_vector_type(8)))  float  v8f;

#define B_  256
#define T_  256
#define E_  256
#define D_  256
#define G4_ (4 * D_)   // 1024

// ---------------------------------------------------------------------------
// fast transcendental helpers
// ---------------------------------------------------------------------------
static __device__ __forceinline__ float fast_tanh(float x) {
#if __has_builtin(__builtin_amdgcn_tanhf)
  return __builtin_amdgcn_tanhf(x);   // V_TANH_F32 on gfx1250
#else
  return tanhf(x);
#endif
}
static __device__ __forceinline__ float fast_sigmoid(float x) {
  return 1.f / (1.f + __expf(-x));
}

// ---------------------------------------------------------------------------
// WMMA fragment helpers (CDNA5 v_wmma_f32_16x16x32_bf16, wave32)
// A 16x32 (MxK), 16-bit: lane<16 -> K in {0..7,16..23}; lane>=16 -> {8..15,24..31}; m = lane%16
// B packed fragments: per lane 16 bf16, same K order as A, n = lane%16
// C/D 16x16 f32: vgpr r -> m = r + (lane>=16 ? 8 : 0), n = lane%16
// ---------------------------------------------------------------------------

static __device__ __forceinline__ v16bf load_a_bf16(const __bf16* A, int lda,
                                                    int m0, int k0, int lane) {
  int m  = m0 + (lane & 15);
  int kb = k0 + ((lane & 16) ? 8 : 0);
  const __bf16* p = A + (size_t)m * lda + kb;
  v8bf lo = *(const v8bf*)p;
  v8bf hi = *(const v8bf*)(p + 16);
  v16bf r;
#pragma unroll
  for (int e = 0; e < 8; ++e) { r[e] = lo[e]; r[8 + e] = hi[e]; }
  return r;
}

static __device__ __forceinline__ v16bf load_a_f32(const float* A, int lda,
                                                   int m0, int k0, int lane) {
  int m  = m0 + (lane & 15);
  int kb = k0 + ((lane & 16) ? 8 : 0);
  const float* p = A + (size_t)m * lda + kb;
  v16bf r;
#pragma unroll
  for (int e = 0; e < 8; ++e) { r[e] = (__bf16)p[e]; r[8 + e] = (__bf16)p[16 + e]; }
  return r;
}

static __device__ __forceinline__ v16bf load_b_frag(const __bf16* Bp, int frag, int lane) {
  return *(const v16bf*)(Bp + ((size_t)frag * 32 + lane) * 16);
}

static __device__ __forceinline__ v8f wmma_bf16(v16bf a, v16bf b, v8f c) {
  return __builtin_amdgcn_wmma_f32_16x16x32_bf16(false, a, false, b, (short)0, c,
                                                 false, false);
}

// ---------------------------------------------------------------------------
// One-time prep kernels
// ---------------------------------------------------------------------------

__global__ void k_cvt_bf16(const float* __restrict__ in, __bf16* __restrict__ out, int n) {
  int i = blockIdx.x * blockDim.x + threadIdx.x;
  if (i < n) out[i] = (__bf16)in[i];
}

// Pack W (N x K over in-features, row stride ldw, column offset col_off) into
// per-lane WMMA B fragments: frag id = nt*KC + kc.
__global__ void k_pack_b(const float* __restrict__ W, int ldw, int col_off,
                         int NT, int KC, __bf16* __restrict__ out) {
  int tid  = blockIdx.x * blockDim.x + threadIdx.x;
  int lane = tid & 31;
  int frag = tid >> 5;
  if (frag >= NT * KC) return;
  int nt = frag / KC, kc = frag % KC;
  int n  = nt * 16 + (lane & 15);
  int kb = kc * 32 + ((lane & 16) ? 8 : 0);
  const float* src = W + (size_t)n * ldw + col_off + kb;
  __bf16* dst = out + ((size_t)frag * 32 + lane) * 16;
#pragma unroll
  for (int e = 0; e < 8; ++e) { dst[e] = (__bf16)src[e]; dst[8 + e] = (__bf16)src[16 + e]; }
}

// W1hcT[j, f] = attn_W1[f, j]  (bf16, j < 2D, f < E) for the fused q-GEMV
__global__ void k_pack_w1hcT(const float* __restrict__ attn_W1, __bf16* __restrict__ out) {
  int idx = blockIdx.x * blockDim.x + threadIdx.x;   // 512*256
  int j = idx >> 8;
  int f = idx & (E_ - 1);
  out[idx] = (__bf16)attn_W1[(size_t)f * (2 * D_ + E_) + j];
}

// enc_projT[b, f, t] = sum_e enc[b,t,e] * W1_e[f,e] + attn_b1[f]   (bf16, TRANSPOSED)
// GEMM M = B*T = 65536, N = E = 256, K = E = 256; epilogue stores [b][f][t].
__global__ void k_encproj(const __bf16* __restrict__ encbf, const __bf16* __restrict__ Bp,
                          const float* __restrict__ b1, __bf16* __restrict__ encprojT) {
  int lane = threadIdx.x & 31;
  int wave = blockIdx.x * (blockDim.x >> 5) + (threadIdx.x >> 5);
  int nt = wave & 15;
  int mt = wave >> 4;
  if (mt >= (B_ * T_) / 16) return;
  const int KC = E_ / 32;
  v8f acc = {};
#pragma unroll
  for (int kc = 0; kc < KC; ++kc) {
    v16bf a = load_a_bf16(encbf, E_, mt * 16, kc * 32, lane);
    v16bf b = load_b_frag(Bp, nt * KC + kc, lane);
    acc = wmma_bf16(a, b, acc);
  }
  int   n    = nt * 16 + (lane & 15);
  float bias = b1[n];
  int   m0   = mt * 16 + ((lane & 16) ? 8 : 0);   // global row = b*T + t
  int   b_i  = m0 >> 8;                           // batch (tile never crosses b)
  int   t0   = m0 & (T_ - 1);                     // 8 consecutive t per lane
  v8bf st;
#pragma unroll
  for (int r = 0; r < 8; ++r) st[r] = (__bf16)(acc[r] + bias);
  *(v8bf*)(encprojT + ((size_t)b_i * E_ + n) * T_ + t0) = st;
}

// ---------------------------------------------------------------------------
// Per-step kernel A: fused q-GEMV + attention scores + softmax + context + y_tilde
// One block per batch element b; 256 threads.
// ---------------------------------------------------------------------------
__global__ void k_attn(const __bf16* __restrict__ encprojT, const __bf16* __restrict__ encbf,
                       const float* __restrict__ h, const float* __restrict__ c,
                       const __bf16* __restrict__ W1hcT,
                       const float* __restrict__ attn_W2, const float* __restrict__ attn_b2,
                       const float* __restrict__ fc_W, const float* __restrict__ fc_b,
                       const float* __restrict__ bias,
                       const float* __restrict__ y_history, int step,
                       float* __restrict__ context, float* __restrict__ y_tilde) {
  __shared__ float shc[2 * D_];   // [h | c] for this batch row
  __shared__ float sq[E_];
  __shared__ float sw2[E_];
  __shared__ float red[T_];
  __shared__ float salpha[T_];
  int b = blockIdx.x;
  int x = threadIdx.x;

  shc[x]       = h[(size_t)b * D_ + x];
  shc[D_ + x]  = c[(size_t)b * D_ + x];
  sw2[x]       = attn_W2[x];
  __syncthreads();

  // q[b, f] = sum_j hc[j] * W1_hc[f, j]  (f = x; coalesced reads of W1hcT, LDS broadcast of hc)
  {
    float qf = 0.f;
#pragma unroll 4
    for (int j = 0; j < 2 * D_; ++j)
      qf += shc[j] * (float)W1hcT[(size_t)j * E_ + x];
    sq[x] = qf;
  }
  __syncthreads();

  // score for timestep t = x: fully lane-coalesced reads of transposed enc_proj
  const __bf16* epT = encprojT + (size_t)b * E_ * T_ + x;
  float s = 0.f;
#pragma unroll 4
  for (int f = 0; f < E_; ++f)
    s += fast_tanh((float)epT[(size_t)f * T_] + sq[f]) * sw2[f];
  s += attn_b2[0];

  // softmax over t (block-wide)
  red[x] = s;
  __syncthreads();
  for (int off = T_ / 2; off > 0; off >>= 1) {
    if (x < off) red[x] = fmaxf(red[x], red[x + off]);
    __syncthreads();
  }
  float mx = red[0];
  __syncthreads();
  float ex = __expf(s - mx);
  red[x] = ex;
  __syncthreads();
  for (int off = T_ / 2; off > 0; off >>= 1) {
    if (x < off) red[x] += red[x + off];
    __syncthreads();
  }
  float inv = 1.f / red[0];
  __syncthreads();
  salpha[x] = ex * inv;
  __syncthreads();

  // context[b, e] with e = x  (coalesced across lanes)
  const __bf16* eb = encbf + (size_t)b * T_ * E_ + x;
  float ctx = 0.f;
#pragma unroll 4
  for (int tt = 0; tt < T_; ++tt)
    ctx += salpha[tt] * (float)eb[(size_t)tt * E_];
  context[(size_t)b * E_ + x] = ctx;

  // y_tilde[b] = context . fc_W[0,:E] + fc_W[0,E]*y_t + fc_b + bias
  red[x] = ctx * fc_W[x];
  __syncthreads();
  for (int off = T_ / 2; off > 0; off >>= 1) {
    if (x < off) red[x] += red[x + off];
    __syncthreads();
  }
  if (x == 0) {
    float yt = y_history[(size_t)b * (T_ - 1) + step];
    y_tilde[b] = red[0] + fc_W[E_] * yt + fc_b[0] + bias[0];
  }
}

// ---------------------------------------------------------------------------
// Per-step kernel B: fused LSTM gates WMMA + pointwise update.
// Each wave owns one (mt, ns) tile of (batch x D) and computes all 4 gate tiles
// (i,f,g,o share the A fragment), then the LSTM update in-register.
// h double-buffered (h_in read, h_out written); c updated in place (unique owner).
// 16*16 = 256 waves -> <<<32, 256>>>.
// ---------------------------------------------------------------------------
__global__ void k_gates_lstm(const float* __restrict__ h_in, const __bf16* __restrict__ Bp,
                             const float* __restrict__ y_tilde, const float* __restrict__ W_ih,
                             const float* __restrict__ b_ih, const float* __restrict__ b_hh,
                             float* __restrict__ h_out, float* __restrict__ c) {
  int lane = threadIdx.x & 31;
  int wave = blockIdx.x * (blockDim.x >> 5) + (threadIdx.x >> 5);
  int ns = wave & 15;        // tile along D (gate column slice)
  int mt = wave >> 4;        // tile along batch
  if (mt >= B_ / 16) return;
  const int KC = D_ / 32;    // 8
  v8f ai = {}, af = {}, ag = {}, ao = {};
#pragma unroll
  for (int kc = 0; kc < KC; ++kc) {
    v16bf a = load_a_f32(h_in, D_, mt * 16, kc * 32, lane);
    ai = wmma_bf16(a, load_b_frag(Bp, (ns +  0) * KC + kc, lane), ai);
    af = wmma_bf16(a, load_b_frag(Bp, (ns + 16) * KC + kc, lane), af);
    ag = wmma_bf16(a, load_b_frag(Bp, (ns + 32) * KC + kc, lane), ag);
    ao = wmma_bf16(a, load_b_frag(Bp, (ns + 48) * KC + kc, lane), ao);
  }
  int n  = ns * 16 + (lane & 15);          // d index
  int m0 = mt * 16 + ((lane & 16) ? 8 : 0);
  float wi = W_ih[n],            bi = b_ih[n]            + b_hh[n];
  float wf = W_ih[D_ + n],       bf = b_ih[D_ + n]       + b_hh[D_ + n];
  float wg = W_ih[2 * D_ + n],   bg = b_ih[2 * D_ + n]   + b_hh[2 * D_ + n];
  float wo = W_ih[3 * D_ + n],   bo = b_ih[3 * D_ + n]   + b_hh[3 * D_ + n];
#pragma unroll
  for (int r = 0; r < 8; ++r) {
    int    m  = m0 + r;
    float  yt = y_tilde[m];
    float  gi = ai[r] + yt * wi + bi;
    float  gf = af[r] + yt * wf + bf;
    float  gg = ag[r] + yt * wg + bg;
    float  go = ao[r] + yt * wo + bo;
    size_t ix = (size_t)m * D_ + n;
    float  cn = fast_sigmoid(gf) * c[ix] + fast_sigmoid(gi) * fast_tanh(gg);
    c[ix]     = cn;
    h_out[ix] = fast_sigmoid(go) * fast_tanh(cn);
  }
}

// ---------------------------------------------------------------------------
// out[b] = (concat(h, context) @ lin3_W.T + lin3_b + bias3) @ fcf_W.T + fcf_b + bias2
// ---------------------------------------------------------------------------
__global__ void k_final(const float* __restrict__ h, const float* __restrict__ context,
                        const float* __restrict__ lin3_W, const float* __restrict__ lin3_b,
                        const float* __restrict__ bias3, const float* __restrict__ fcf_W,
                        const float* __restrict__ fcf_b, const float* __restrict__ bias2,
                        float* __restrict__ out) {
  __shared__ float red[D_];
  int b = blockIdx.x;
  int d = threadIdx.x;
  const float* w = lin3_W + (size_t)d * (D_ + E_);
  float p = lin3_b[d] + bias3[d];
#pragma unroll 4
  for (int j = 0; j < D_; ++j) p += h[(size_t)b * D_ + j] * w[j];
#pragma unroll 4
  for (int j = 0; j < E_; ++j) p += context[(size_t)b * E_ + j] * w[D_ + j];
  red[d] = p * fcf_W[d];
  __syncthreads();
  for (int off = D_ / 2; off > 0; off >>= 1) {
    if (d < off) red[d] += red[d + off];
    __syncthreads();
  }
  if (d == 0) out[b] = red[0] + fcf_b[0] + bias2[0];
}

// ---------------------------------------------------------------------------
// Host orchestration
// ---------------------------------------------------------------------------

extern "C" void kernel_launch(void* const* d_in, const int* in_sizes, int n_in,
                              void* d_out, int out_size, void* d_ws, size_t ws_size,
                              hipStream_t stream) {
  const float* input_encoded = (const float*)d_in[0];
  const float* y_history     = (const float*)d_in[1];
  const float* attn_W1       = (const float*)d_in[2];   // (E, 2D+E), ldw = 768
  const float* attn_b1       = (const float*)d_in[3];
  const float* attn_W2       = (const float*)d_in[4];
  const float* attn_b2       = (const float*)d_in[5];
  const float* fc_W          = (const float*)d_in[6];
  const float* fc_b          = (const float*)d_in[7];
  const float* bias          = (const float*)d_in[8];
  const float* W_ih          = (const float*)d_in[9];
  const float* W_hh          = (const float*)d_in[10];  // (4D, D)
  const float* b_ih          = (const float*)d_in[11];
  const float* b_hh          = (const float*)d_in[12];
  const float* lin3_W        = (const float*)d_in[13];
  const float* lin3_b        = (const float*)d_in[14];
  const float* bias3         = (const float*)d_in[15];
  const float* fcf_W         = (const float*)d_in[16];
  const float* fcf_b         = (const float*)d_in[17];
  const float* bias2         = (const float*)d_in[18];

  // workspace layout (256B-aligned slabs)
  char*  base = (char*)d_ws;
  size_t off  = 0;
  auto alloc = [&](size_t bytes) -> char* {
    char* p = base + off;
    off += (bytes + 255) & ~(size_t)255;
    return p;
  };
  __bf16* encbf    = (__bf16*)alloc((size_t)B_ * T_ * E_ * 2);   // 33.5 MB
  __bf16* encprojT = (__bf16*)alloc((size_t)B_ * T_ * E_ * 2);   // 33.5 MB (transposed)
  __bf16* BpW1e    = (__bf16*)alloc((size_t)E_ * E_ * 2);        // 128 KB
  __bf16* BpWhh    = (__bf16*)alloc((size_t)G4_ * D_ * 2);       // 512 KB
  __bf16* W1hcT    = (__bf16*)alloc((size_t)2 * D_ * E_ * 2);    // 256 KB
  float*  context  = (float*)alloc((size_t)B_ * E_ * 4);
  float*  y_tilde  = (float*)alloc((size_t)B_ * 4);
  float*  hA       = (float*)alloc((size_t)B_ * D_ * 4);
  float*  hB       = (float*)alloc((size_t)B_ * D_ * 4);
  float*  cbuf     = (float*)alloc((size_t)B_ * D_ * 4);
  (void)ws_size; (void)in_sizes; (void)n_in; (void)out_size;

  // h0 = c0 = 0 (hB is fully written by step 0 before any read)
  hipMemsetAsync(hA,   0, (size_t)B_ * D_ * 4, stream);
  hipMemsetAsync(cbuf, 0, (size_t)B_ * D_ * 4, stream);

  // one-time prep
  {
    int n = B_ * T_ * E_;
    k_cvt_bf16<<<(n + 255) / 256, 256, 0, stream>>>(input_encoded, encbf, n);
  }
  // W1_e = attn_W1[:, 2D:] -> B-fragments (N=256, K=256)
  k_pack_b<<<(16 * 8 * 32 + 255) / 256, 256, 0, stream>>>(attn_W1, 2 * D_ + E_, 2 * D_, 16, 8, BpW1e);
  // W_hh -> B-fragments (N=1024, K=256)
  k_pack_b<<<(64 * 8 * 32 + 255) / 256, 256, 0, stream>>>(W_hh, D_, 0, 64, 8, BpWhh);
  // W1_hc transposed bf16 for the fused q-GEMV
  k_pack_w1hcT<<<(2 * D_ * E_) / 256, 256, 0, stream>>>(attn_W1, W1hcT);

  {
    int waves  = ((B_ * T_) / 16) * (E_ / 16);   // 65536
    int blocks = (waves * 32) / 256;             // 8192
    k_encproj<<<blocks, 256, 0, stream>>>(encbf, BpW1e, attn_b1, encprojT);
  }

  // recurrent scan: T-1 = 255 steps, 2 kernels per step
  float* hcur = hA;
  float* hnxt = hB;
  for (int s = 0; s < T_ - 1; ++s) {
    k_attn<<<B_, T_, 0, stream>>>(encprojT, encbf, hcur, cbuf, W1hcT,
                                  attn_W2, attn_b2, fc_W, fc_b, bias,
                                  y_history, s, context, y_tilde);
    k_gates_lstm<<<(16 * 16 * 32) / 256, 256, 0, stream>>>(hcur, BpWhh, y_tilde,
                                                           W_ih, b_ih, b_hh, hnxt, cbuf);
    float* tmp = hcur; hcur = hnxt; hnxt = tmp;
  }

  k_final<<<B_, D_, 0, stream>>>(hcur, context, lin3_W, lin3_b, bias3,
                                 fcf_W, fcf_b, bias2, (float*)d_out);
}
